// InteractionBlock_20779051778082
// MI455X (gfx1250) — matrix operand order repaired and checked by reading, not compiled, not run on hardware
//
#include <hip/hip_runtime.h>
#include <hip/hip_bf16.h>

// Problem constants (match reference)
#define NN    10000
#define EE    320000
#define HIDD  256
#define NRBFD 64
#define NFD   256

typedef __attribute__((ext_vector_type(16))) __bf16 v16bf;
typedef __attribute__((ext_vector_type(8)))  __bf16 v8bf;
typedef __attribute__((ext_vector_type(8)))  float  v8f;

// ---------------------------------------------------------------------------
// WMMA helper: D = A(16x32 bf16) * B(32x16 bf16) + C(16x16 f32)
// ---------------------------------------------------------------------------
__device__ __forceinline__ v8f wmma_bf16(v16bf a, v16bf b, v8f c) {
  return __builtin_amdgcn_wmma_f32_16x16x32_bf16(
      /*neg_a=*/false, a, /*neg_b=*/false, b,
      /*c_mod=*/(short)0, c, /*reuse_a=*/false, /*reuse_b=*/false);
}

// A-fragment (16x32 bf16) from an f32 row-major source, converting on the fly.
// Lane holds row (lane%16); elements 0..7 = K[kb+8*half .. +7],
// elements 8..15 = K[kb+16+8*half .. +7].
__device__ __forceinline__ v16bf load_afrag_f32(const float* rowPtr, int kb, int half) {
  const v8f lo = *(const v8f*)(rowPtr + kb + 8 * half);
  const v8f hi = *(const v8f*)(rowPtr + kb + 16 + 8 * half);
  v16bf a;
#pragma unroll
  for (int i = 0; i < 8; ++i) {
    a[i]     = (__bf16)lo[i];
    a[i + 8] = (__bf16)hi[i];
  }
  return a;
}

// A-fragment from a bf16 LDS row (row base already selected for lane%16).
__device__ __forceinline__ v16bf load_afrag_lds(const __bf16* row, int kb, int half) {
  v8bf lo = *(const v8bf*)(row + kb + 8 * half);
  v8bf hi = *(const v8bf*)(row + kb + 16 + 8 * half);
  return __builtin_shufflevector(lo, hi, 0, 1, 2, 3, 4, 5, 6, 7,
                                          8, 9, 10, 11, 12, 13, 14, 15);
}

// B-fragment (32x16 bf16) from transposed bf16 weights WT[N][K] (row-major):
// elements 0..7 = W[kb + 8*half + i][col], elements 8..15 = W[kb + 16 + 8*half + (i-8)][col].
__device__ __forceinline__ v16bf load_bfrag(const __bf16* WT, int Kdim, int col,
                                            int kb, int half) {
  const __bf16* p = WT + (size_t)col * Kdim + kb;
  v8bf lo = *(const v8bf*)(p + 8 * half);
  v8bf hi = *(const v8bf*)(p + 16 + 8 * half);
  return __builtin_shufflevector(lo, hi, 0, 1, 2, 3, 4, 5, 6, 7,
                                          8, 9, 10, 11, 12, 13, 14, 15);
}

__device__ __forceinline__ float silu(float v) {
  return v / (1.0f + __expf(-v));
}

// ---------------------------------------------------------------------------
// Kernel 0: zero scatter accumulator + transpose/convert weights to bf16
// ---------------------------------------------------------------------------
__global__ __launch_bounds__(256) void k_prep(
    const float* __restrict__ Wm1, const float* __restrict__ Wm2,
    const float* __restrict__ Wl1, const float* __restrict__ Wl2,
    const float* __restrict__ Wl,
    __bf16* __restrict__ WT1, __bf16* __restrict__ WT2,
    __bf16* __restrict__ WTl1, __bf16* __restrict__ WTl2,
    __bf16* __restrict__ WTl, float* __restrict__ agg) {
  const int stride = gridDim.x * blockDim.x;
  const int tid = blockIdx.x * blockDim.x + threadIdx.x;
  // W_mlp1 [64,256] -> WT1 [256][64]
  for (int i = tid; i < NRBFD * NFD; i += stride) {
    int k = i >> 8, n = i & 255;
    WT1[n * NRBFD + k] = (__bf16)Wm1[i];
  }
  // 256x256 weights -> transposed bf16
  for (int i = tid; i < NFD * NFD; i += stride) {
    int k = i >> 8, n = i & 255;
    WT2[n * NFD + k]  = (__bf16)Wm2[i];
    WTl1[n * NFD + k] = (__bf16)Wl1[i];
    WTl2[n * NFD + k] = (__bf16)Wl2[i];
    WTl[n * NFD + k]  = (__bf16)Wl[i];
  }
  // zero agg (required every launch: atomics accumulate into it)
  for (int i = tid; i < NN * NFD; i += stride) agg[i] = 0.0f;
}

// ---------------------------------------------------------------------------
// Kernel 1: xf = x @ W_lin1   (per wave: one 16-row tile, all 16 col tiles)
// ---------------------------------------------------------------------------
__global__ __launch_bounds__(256, 1) void k_lin1(
    const float* __restrict__ x, const __bf16* __restrict__ WTl1,
    float* __restrict__ xf, int nTiles) {
  const int wid = threadIdx.x >> 5;
  const int lane = threadIdx.x & 31;
  const int tile = blockIdx.x * 8 + wid;
  if (tile >= nTiles) return;
  const int half = lane >> 4;
  const int n = lane & 15;

  const float* rowPtr = x + (size_t)(tile * 16 + n) * HIDD;
  v16bf afr[8];
#pragma unroll
  for (int kt = 0; kt < 8; ++kt) afr[kt] = load_afrag_f32(rowPtr, 32 * kt, half);

  for (int j = 0; j < 16; ++j) {
    v8f acc = {};
#pragma unroll
    for (int kt = 0; kt < 8; ++kt)
      acc = wmma_bf16(afr[kt], load_bfrag(WTl1, HIDD, j * 16 + n, 32 * kt, half), acc);
#pragma unroll
    for (int r = 0; r < 8; ++r)
      xf[(size_t)(tile * 16 + r + 8 * half) * NFD + j * 16 + n] = acc[r];
  }
}

// ---------------------------------------------------------------------------
// Kernel 2: fused edge pipeline. Per wave = 16 edges:
//   h = silu(edge_attr @ W_mlp1 + b1)            (GEMM1, K=64, h -> LDS bf16)
//   W = (h @ W_mlp2 + b2) * cutoff(edge_weight)  (GEMM2, K=256)
//   agg[dst] += xf[src] * W                      (gather + f32 atomics)
// ---------------------------------------------------------------------------
#define EW 4  // waves per block (keeps static LDS at 32KB + meta)
__global__ __launch_bounds__(EW * 32, 1) void k_edge(
    const float* __restrict__ edge_attr, const long long* __restrict__ eidx,
    const float* __restrict__ eweight, const float* __restrict__ bm1,
    const float* __restrict__ bm2, const __bf16* __restrict__ WT1,
    const __bf16* __restrict__ WT2, const float* __restrict__ xf,
    float* __restrict__ agg, int nTiles) {
  __shared__ __align__(32) __bf16 hlds[EW][16 * NFD];
  __shared__ int   srcLds[EW][16];
  __shared__ int   dstLds[EW][16];
  __shared__ float cutLds[EW][16];

  const int wid = threadIdx.x >> 5;
  const int lane = threadIdx.x & 31;
  const int tile = blockIdx.x * EW + wid;
  if (tile >= nTiles) return;
  const int half = lane >> 4;
  const int n = lane & 15;
  const int eBase = tile * 16;

  if (lane < 16) {
    const int e = eBase + lane;
    srcLds[wid][lane] = (int)eidx[e];
    dstLds[wid][lane] = (int)eidx[EE + e];
    float d = eweight[e];
    float c = 0.5f * (__cosf(d * 0.6283185307f) + 1.0f);  // pi/5
    cutLds[wid][lane] = (d < 5.0f) ? c : 0.0f;
  }

  // ---- GEMM1: h = silu(edge_attr @ W_mlp1 + b1) ----
  const float* arow = edge_attr + (size_t)(eBase + n) * NRBFD;
  const v16bf a0 = load_afrag_f32(arow, 0, half);
  const v16bf a1 = load_afrag_f32(arow, 32, half);

  for (int j = 0; j < 16; ++j) {
    v8f acc = {};
    acc = wmma_bf16(a0, load_bfrag(WT1, NRBFD, j * 16 + n, 0, half), acc);
    acc = wmma_bf16(a1, load_bfrag(WT1, NRBFD, j * 16 + n, 32, half), acc);
    const float bias = bm1[j * 16 + n];
#pragma unroll
    for (int r = 0; r < 8; ++r) {
      float v = silu(acc[r] + bias);
      hlds[wid][(r + 8 * half) * NFD + j * 16 + n] = (__bf16)v;
    }
  }

  // per-lane edge metadata for the 8 accumulator rows (same-wave LDS, in order)
  float cutv[8];
  int srcv[8], dstv[8];
#pragma unroll
  for (int r = 0; r < 8; ++r) {
    cutv[r] = cutLds[wid][8 * half + r];
    srcv[r] = srcLds[wid][8 * half + r];
    dstv[r] = dstLds[wid][8 * half + r];
  }

  // ---- GEMM2 + cutoff + gather-multiply-scatter ----
  const __bf16* hrow = &hlds[wid][n * NFD];
  for (int j = 0; j < 16; ++j) {
    v8f acc = {};
#pragma unroll
    for (int kt = 0; kt < 8; ++kt) {
      v16bf a = load_afrag_lds(hrow, 32 * kt, half);
      v16bf b = load_bfrag(WT2, NFD, j * 16 + n, 32 * kt, half);
      acc = wmma_bf16(a, b, acc);
    }
    const int col = j * 16 + n;
    const float bias = bm2[col];
#pragma unroll
    for (int r = 0; r < 8; ++r) {
      float w = (acc[r] + bias) * cutv[r];
      float msg = xf[(size_t)srcv[r] * NFD + col] * w;
      atomicAdd(&agg[(size_t)dstv[r] * NFD + col], msg);
    }
  }
}

// ---------------------------------------------------------------------------
// Kernel 3: out = silu(agg @ W_lin2 + b_lin2) @ W_lin + b_lin
// ---------------------------------------------------------------------------
#define TW 4
__global__ __launch_bounds__(TW * 32, 1) void k_tail(
    const float* __restrict__ agg, const __bf16* __restrict__ WTl2,
    const float* __restrict__ bl2, const __bf16* __restrict__ WTl,
    const float* __restrict__ bl, float* __restrict__ out, int nTiles) {
  __shared__ __align__(32) __bf16 hlds[TW][16 * NFD];
  const int wid = threadIdx.x >> 5;
  const int lane = threadIdx.x & 31;
  const int tile = blockIdx.x * TW + wid;
  if (tile >= nTiles) return;
  const int half = lane >> 4;
  const int n = lane & 15;

  const float* rowPtr = agg + (size_t)(tile * 16 + n) * NFD;
  v16bf afr[8];
#pragma unroll
  for (int kt = 0; kt < 8; ++kt) afr[kt] = load_afrag_f32(rowPtr, 32 * kt, half);

  // GEMM1: y = silu(agg @ W_lin2 + b_lin2) -> LDS bf16
  for (int j = 0; j < 16; ++j) {
    v8f acc = {};
#pragma unroll
    for (int kt = 0; kt < 8; ++kt)
      acc = wmma_bf16(afr[kt], load_bfrag(WTl2, NFD, j * 16 + n, 32 * kt, half), acc);
    const float bias = bl2[j * 16 + n];
#pragma unroll
    for (int r = 0; r < 8; ++r) {
      float v = silu(acc[r] + bias);
      hlds[wid][(r + 8 * half) * NFD + j * 16 + n] = (__bf16)v;
    }
  }

  // GEMM2: out = y @ W_lin + b_lin
  const __bf16* hrow = &hlds[wid][n * NFD];
  for (int j = 0; j < 16; ++j) {
    v8f acc = {};
#pragma unroll
    for (int kt = 0; kt < 8; ++kt) {
      v16bf a = load_afrag_lds(hrow, 32 * kt, half);
      v16bf b = load_bfrag(WTl, NFD, j * 16 + n, 32 * kt, half);
      acc = wmma_bf16(a, b, acc);
    }
    const int col = j * 16 + n;
    const float bias = bl[col];
#pragma unroll
    for (int r = 0; r < 8; ++r)
      out[(size_t)(tile * 16 + r + 8 * half) * HIDD + col] = acc[r] + bias;
  }
}

// ---------------------------------------------------------------------------
// Host launcher
// ---------------------------------------------------------------------------
extern "C" void kernel_launch(void* const* d_in, const int* in_sizes, int n_in,
                              void* d_out, int out_size, void* d_ws, size_t ws_size,
                              hipStream_t stream) {
  (void)in_sizes; (void)n_in; (void)out_size; (void)ws_size;

  const float*     x    = (const float*)d_in[0];
  const long long* eidx = (const long long*)d_in[1];   // int64 edge_index [2,E]
  const float*     ew   = (const float*)d_in[2];
  const float*     ea   = (const float*)d_in[3];
  const float*     Wm1  = (const float*)d_in[4];
  const float*     bm1  = (const float*)d_in[5];
  const float*     Wm2  = (const float*)d_in[6];
  const float*     bm2  = (const float*)d_in[7];
  const float*     Wl1  = (const float*)d_in[8];
  const float*     Wl2  = (const float*)d_in[9];
  const float*     bl2  = (const float*)d_in[10];
  const float*     Wl   = (const float*)d_in[11];
  const float*     bl   = (const float*)d_in[12];
  float*           out  = (float*)d_out;

  // Workspace layout (bytes)
  char* ws = (char*)d_ws;
  const size_t AGG_B = (size_t)NN * NFD * sizeof(float);   // 10,240,000
  const size_t XF_B  = (size_t)NN * NFD * sizeof(float);   // 10,240,000
  const size_t WT1_B = (size_t)NFD * NRBFD * sizeof(__bf16);
  const size_t WTQ_B = (size_t)NFD * NFD * sizeof(__bf16);
  float*  agg  = (float*)ws;
  float*  xf   = (float*)(ws + AGG_B);
  __bf16* WT1  = (__bf16*)(ws + AGG_B + XF_B);
  __bf16* WT2  = (__bf16*)(ws + AGG_B + XF_B + WT1_B);
  __bf16* WTl1 = (__bf16*)(ws + AGG_B + XF_B + WT1_B + 1 * WTQ_B);
  __bf16* WTl2 = (__bf16*)(ws + AGG_B + XF_B + WT1_B + 2 * WTQ_B);
  __bf16* WTlf = (__bf16*)(ws + AGG_B + XF_B + WT1_B + 3 * WTQ_B);

  // 0) zero accumulator + build bf16 transposed weights
  k_prep<<<1024, 256, 0, stream>>>(Wm1, Wm2, Wl1, Wl2, Wl,
                                   WT1, WT2, WTl1, WTl2, WTlf, agg);

  // 1) xf = x @ W_lin1 ; 625 row-tiles, 8 waves/block
  k_lin1<<<(625 + 7) / 8, 256, 0, stream>>>(x, WTl1, xf, 625);

  // 2) fused edge MLP + CFConv scatter ; 20000 edge tiles, 4 waves/block
  k_edge<<<(20000 + EW - 1) / EW, EW * 32, 0, stream>>>(
      ea, eidx, ew, bm1, bm2, WT1, WT2, xf, agg, 20000);

  // 3) tail: silu(agg @ W_lin2 + b) @ W_lin + b ; 625 tiles, 4 waves/block
  k_tail<<<(625 + TW - 1) / TW, TW * 32, 0, stream>>>(
      agg, WTl2, bl2, WTlf, bl, out, 625);
}
